// LinkDecoder_52132313039373
// MI455X (gfx1250) — compile-verified, matrix operand order checked by reading, not run on hardware
//
#include <hip/hip_runtime.h>
#include <stdint.h>

#define DIN   256
#define K_TOT 512          // 2*DIN
#define TILE_M 32          // edges per block (2 WMMA M-tiles)
#define BLOCK_THREADS 256

typedef __attribute__((ext_vector_type(16))) __bf16        v16bf;
typedef __attribute__((ext_vector_type(8)))  float         v8f;
typedef __attribute__((ext_vector_type(8)))  unsigned short ushort8;
typedef __attribute__((ext_vector_type(4)))  float         float4v;

union FragU { struct { ushort8 lo; ushort8 hi; } s; v16bf v; };

__device__ __forceinline__ unsigned short f32_to_bf16(float f) {
  unsigned u = __builtin_bit_cast(unsigned, f);
  u += 0x7FFFu + ((u >> 16) & 1u);          // round-to-nearest-even
  return (unsigned short)(u >> 16);
}

// ---------------------------------------------------------------------------
// Repack W1[512,256] (fp32 row-major, k-major) into bf16 WMMA B-fragments.
// Fragment (nt,kt): 32 lanes x 16 bf16, each lane's 16 values contiguous.
// Lane L -> column n = nt*16 + (L&15); K half-select base = (L<16)?0:8.
// Element i: j=i>>1, bit=i&1; k_local = j<4 ? base+2j+bit : 16+base+2(j-4)+bit.
// ---------------------------------------------------------------------------
__global__ void repack_w1_kernel(const float* __restrict__ W1,
                                 unsigned short* __restrict__ wB) {
  int t = blockIdx.x * blockDim.x + threadIdx.x;   // 16*16*32 = 8192 threads
  if (t >= 16 * 16 * 32) return;
  int lane = t & 31;
  int kt   = (t >> 5) & 15;
  int nt   = t >> 9;
  int n    = nt * 16 + (lane & 15);
  int base = (lane < 16) ? 0 : 8;
  unsigned short* dst = wB + (size_t)t * 16;
#pragma unroll
  for (int i = 0; i < 16; ++i) {
    int j = i >> 1, bit = i & 1;
    int kl = (j < 4) ? (base + 2 * j + bit) : (16 + base + 2 * (j - 4) + bit);
    int k  = kt * 32 + kl;
    dst[i] = f32_to_bf16(W1[k * DIN + n]);
  }
}

// ---------------------------------------------------------------------------
// Main kernel: 32 edges per block, 8 waves.
// Wave w owns output cols [32w, 32w+32) x all 32 rows: 2x2 WMMA register tile.
// K-loop software-pipelined: fragments for kt+1 are fetched before the WMMAs
// of kt issue, giving one full iteration of load-latency hiding per wave.
// ---------------------------------------------------------------------------
__global__ __launch_bounds__(BLOCK_THREADS)
void link_decoder_kernel(const float* __restrict__ x,
                         const int64_t* __restrict__ eidx,
                         const unsigned short* __restrict__ wB,
                         const float* __restrict__ b1,
                         const float* __restrict__ W2,
                         const float* __restrict__ b2,
                         float* __restrict__ out,
                         int E) {
  __shared__ __align__(16) unsigned short hsh[TILE_M][K_TOT + 8]; // bf16 concat rows
  __shared__ float   h1[TILE_M][DIN + 4];                         // relu(hW1+b1)
  __shared__ float   p2[TILE_M][8];
  __shared__ int64_t sidx[TILE_M], didx[TILE_M];

  const int tid  = threadIdx.x;
  const int lane = tid & 31;
  const int wave = tid >> 5;
  const int e0   = blockIdx.x * TILE_M;

  if (tid < TILE_M) {
    int e = e0 + tid; if (e >= E) e = E - 1;
    sidx[tid] = eidx[e];
  } else if (tid < 2 * TILE_M) {
    int r = tid - TILE_M;
    int e = e0 + r; if (e >= E) e = E - 1;
    didx[r] = eidx[(size_t)E + e];
  }
  __syncthreads();

  // Gather + fp32->bf16, float4-vectorized, coalesced per row segment.
  for (int idx = tid; idx < TILE_M * (K_TOT / 4); idx += BLOCK_THREADS) {
    int row = idx >> 7;                 // K_TOT/4 == 128
    int c4  = (idx & 127) * 4;          // col in [0,512)
    int64_t node = (c4 < DIN) ? sidx[row] : didx[row];
    int cc = c4 & (DIN - 1);
    float4v v = *(const float4v*)(x + (size_t)node * DIN + cc);
    hsh[row][c4 + 0] = f32_to_bf16(v[0]);
    hsh[row][c4 + 1] = f32_to_bf16(v[1]);
    hsh[row][c4 + 2] = f32_to_bf16(v[2]);
    hsh[row][c4 + 3] = f32_to_bf16(v[3]);
  }
  __syncthreads();

  // Layer 1: [32,512]x[512,256] via v_wmma_f32_16x16x32_bf16, 2x2 tile/wave.
  const int nt0  = wave * 2;
  const int rowA = lane & 15;           // A-fragment row within M-tile
  const int kb   = (lane & 16) ? 8 : 0; // A-fragment K half-select
  const unsigned short* wB0 = wB + ((size_t)(nt0 * 16) * 32 + lane) * 16;
  const unsigned short* wB1 = wB + ((size_t)((nt0 + 1) * 16) * 32 + lane) * 16;

  v8f c00 = {}, c01 = {}, c10 = {}, c11 = {};

  FragU a0c, a1c, b0c, b1c;
  // Prologue: fetch fragments for kt = 0.
  a0c.s.lo = *(const ushort8*)(&hsh[rowA][kb]);
  a0c.s.hi = *(const ushort8*)(&hsh[rowA][16 + kb]);
  a1c.s.lo = *(const ushort8*)(&hsh[rowA + 16][kb]);
  a1c.s.hi = *(const ushort8*)(&hsh[rowA + 16][16 + kb]);
  {
    const ushort8* p0 = (const ushort8*)wB0;
    const ushort8* p1 = (const ushort8*)wB1;
    b0c.s.lo = p0[0]; b0c.s.hi = p0[1];
    b1c.s.lo = p1[0]; b1c.s.hi = p1[1];
  }

#pragma unroll
  for (int kt = 0; kt < 16; ++kt) {
    FragU a0n, a1n, b0n, b1n;
    if (kt < 15) {
      int k0 = (kt + 1) * 32;
      a0n.s.lo = *(const ushort8*)(&hsh[rowA][k0 + kb]);
      a0n.s.hi = *(const ushort8*)(&hsh[rowA][k0 + 16 + kb]);
      a1n.s.lo = *(const ushort8*)(&hsh[rowA + 16][k0 + kb]);
      a1n.s.hi = *(const ushort8*)(&hsh[rowA + 16][k0 + 16 + kb]);
      const ushort8* p0 = (const ushort8*)(wB0 + (size_t)(kt + 1) * 32 * 16);
      const ushort8* p1 = (const ushort8*)(wB1 + (size_t)(kt + 1) * 32 * 16);
      b0n.s.lo = p0[0]; b0n.s.hi = p0[1];
      b1n.s.lo = p1[0]; b1n.s.hi = p1[1];
    }
    c00 = __builtin_amdgcn_wmma_f32_16x16x32_bf16(false, a0c.v, false, b0c.v,
                                                  (short)0, c00, false, false);
    c01 = __builtin_amdgcn_wmma_f32_16x16x32_bf16(false, a0c.v, false, b1c.v,
                                                  (short)0, c01, false, false);
    c10 = __builtin_amdgcn_wmma_f32_16x16x32_bf16(false, a1c.v, false, b0c.v,
                                                  (short)0, c10, false, false);
    c11 = __builtin_amdgcn_wmma_f32_16x16x32_bf16(false, a1c.v, false, b1c.v,
                                                  (short)0, c11, false, false);
    if (kt < 15) {
      a0c = a0n; a1c = a1n; b0c = b0n; b1c = b1n;
    }
  }

  // Bias + ReLU from C-fragment layout: VGPR r -> M = r + 8*(lane>=16), N = lane&15.
  {
    int m_off = (lane & 16) ? 8 : 0;
    int n0 = wave * 32 + (lane & 15);
    int n1 = n0 + 16;
    float bb0 = b1[n0], bb1 = b1[n1];
#pragma unroll
    for (int r = 0; r < 8; ++r) {
      int m = r + m_off;
      float v00 = c00[r] + bb0; h1[m][n0]      = v00 > 0.f ? v00 : 0.f;
      float v01 = c01[r] + bb1; h1[m][n1]      = v01 > 0.f ? v01 : 0.f;
      float v10 = c10[r] + bb0; h1[m + 16][n0] = v10 > 0.f ? v10 : 0.f;
      float v11 = c11[r] + bb1; h1[m + 16][n1] = v11 > 0.f ? v11 : 0.f;
    }
  }
  __syncthreads();

  // Layer 2: out[e] = dot(h1[row], W2) + b2 ; 8 threads per edge + tree sum.
  {
    int r2  = tid >> 3;        // 32 rows
    int seg = tid & 7;         // 8 segments of 32 cols
    float s = 0.f;
#pragma unroll
    for (int i = 0; i < 32; ++i) {
      int c = seg * 32 + i;
      s += h1[r2][c] * W2[c];
    }
    p2[r2][seg] = s;
  }
  __syncthreads();
  if (tid < TILE_M) {
    float s = 0.f;
#pragma unroll
    for (int i = 0; i < 8; ++i) s += p2[tid][i];
    int e = e0 + tid;
    if (e < E) out[e] = s + b2[0];
  }
}

extern "C" void kernel_launch(void* const* d_in, const int* in_sizes, int n_in,
                              void* d_out, int out_size, void* d_ws, size_t ws_size,
                              hipStream_t stream) {
  const float*   x    = (const float*)d_in[0];
  const int64_t* eidx = (const int64_t*)d_in[1];   // int64 [2,E]
  const float*   W1   = (const float*)d_in[2];
  const float*   b1   = (const float*)d_in[3];
  const float*   W2   = (const float*)d_in[4];
  const float*   b2   = (const float*)d_in[5];
  float* out = (float*)d_out;

  int E = in_sizes[1] / 2;
  unsigned short* wB = (unsigned short*)d_ws;      // 256 KB repacked bf16 W1

  repack_w1_kernel<<<32, 256, 0, stream>>>(W1, wB);
  int nblocks = (E + TILE_M - 1) / TILE_M;
  link_decoder_kernel<<<nblocks, BLOCK_THREADS, 0, stream>>>(
      x, eidx, wB, b1, W2, b2, out, E);
}
